// SparseAutoencoder_39298950758727
// MI455X (gfx1250) — matrix operand order repaired and checked by reading, not compiled, not run on hardware
//
#include <hip/hip_runtime.h>
#include <hip/hip_bf16.h>

// ---------------------------------------------------------------------------
// Sparse autoencoder forward for MI455X (gfx1250, wave32, WMMA).
//   z     = x @ W_enc.T            [8192,16384]  (f16 hi/lo split WMMA)
//   z_s   = topk(z, 50)            (radix select per row)
//   recon = z_s @ W_dec.T          (sparse gather of 50 W_dec columns/row)
// d_out = recon[8192*1024] ++ z_sparse[8192*16384]   (float32)
//
// The f32 -> f16 hi/lo decomposition is done ONCE by a prep kernel into
// workspace, so the GEMM hot loop is pure b128 loads + v_wmma (no cvt VALU).
// ---------------------------------------------------------------------------

#define B_ROWS 8192
#define K_DIM  1024
#define H_DIM  16384
#define TOPK   50

typedef __attribute__((ext_vector_type(16))) _Float16 v16h;
typedef __attribute__((ext_vector_type(8)))  _Float16 v8h;
typedef __attribute__((ext_vector_type(8)))  float    v8f;

// order-preserving float -> uint key (larger float => larger key)
__device__ __forceinline__ unsigned ord_key(float v) {
    unsigned u = __float_as_uint(v);
    return (u & 0x80000000u) ? ~u : (u | 0x80000000u);
}

// ---------------------------------------------------------------------------
// One-shot f32 -> f16 (hi, lo) split:  f ~= hi + lo with ~2^-22 rel error.
// Each thread handles 8 consecutive elements (float4 in, v8h out).
// ---------------------------------------------------------------------------
__global__ __launch_bounds__(256) void sae_split_f16(
        const float* __restrict__ src, _Float16* __restrict__ hi,
        _Float16* __restrict__ lo, int n8) {
    const int i = blockIdx.x * 256 + threadIdx.x;
    if (i >= n8) return;
    const float4 u0 = *(const float4*)(src + (size_t)i * 8);
    const float4 u1 = *(const float4*)(src + (size_t)i * 8 + 4);
    const float f[8] = {u0.x, u0.y, u0.z, u0.w, u1.x, u1.y, u1.z, u1.w};
    v8h h, l;
#pragma unroll
    for (int e = 0; e < 8; ++e) {
        _Float16 hh = (_Float16)f[e];
        h[e] = hh;
        l[e] = (_Float16)(f[e] - (float)hh);
    }
    *(v8h*)(hi + (size_t)i * 8) = h;
    *(v8h*)(lo + (size_t)i * 8) = l;
}

// ---------------------------------------------------------------------------
// Encoder GEMM: z[b,h] = dot(x[b,:], W_enc[h,:]).  Both operands K-contiguous
// => direct global->fragment f16 loads matching the wave32 WMMA layouts.
// Each wave computes a 64x64 tile: 4 M-frags x 4 N-frags, K stepped by 32.
// D += Ahi*Bhi + Ahi*Blo + Alo*Bhi   (Alo*Blo ~ 2^-22, dropped).
// ---------------------------------------------------------------------------
__global__ __launch_bounds__(256) void sae_enc_gemm(
        const _Float16* __restrict__ xh, const _Float16* __restrict__ xl,
        const _Float16* __restrict__ wh, const _Float16* __restrict__ wl,
        float* __restrict__ z) {
    const int lane = threadIdx.x & 31;
    const int wave = threadIdx.x >> 5;
    const int wid  = blockIdx.x * 8 + wave;
    const int NT   = H_DIM / 64;                 // 256 n-tiles
    const int m0   = (wid / NT) * 64;
    const int n0   = (wid % NT) * 64;
    const int hx   = lane >> 4;                  // half-wave select
    const int l15  = lane & 15;

    v8f acc[4][4] = {};

    for (int k0 = 0; k0 < K_DIM; k0 += 32) {
        // A frags: lane l15 = row m; elems[0..7]=K[hx*8..+7], [8..15]=K[16+hx*8..+7]
        v16h ahi[4], alo[4];
#pragma unroll
        for (int mi = 0; mi < 4; ++mi) {
            const size_t base = (size_t)(m0 + mi * 16 + l15) * K_DIM + k0 + hx * 8;
            v8h h0 = *(const v8h*)(xh + base);
            v8h h1 = *(const v8h*)(xh + base + 16);
            v8h l0 = *(const v8h*)(xl + base);
            v8h l1 = *(const v8h*)(xl + base + 16);
            ahi[mi] = __builtin_shufflevector(h0, h1, 0, 1, 2, 3, 4, 5, 6, 7,
                                              8, 9, 10, 11, 12, 13, 14, 15);
            alo[mi] = __builtin_shufflevector(l0, l1, 0, 1, 2, 3, 4, 5, 6, 7,
                                              8, 9, 10, 11, 12, 13, 14, 15);
        }
#pragma unroll
        for (int ni = 0; ni < 4; ++ni) {
            // B frags: lane l15 = column n (= W_enc row); elems[0..7]=K[hx*8..+7], [8..15]=K[16+hx*8..+7]
            const size_t bb = (size_t)(n0 + ni * 16 + l15) * K_DIM + k0 + hx * 8;
            v16h bhi = __builtin_shufflevector(*(const v8h*)(wh + bb), *(const v8h*)(wh + bb + 16),
                                               0, 1, 2, 3, 4, 5, 6, 7, 8, 9, 10, 11, 12, 13, 14, 15);
            v16h blo = __builtin_shufflevector(*(const v8h*)(wl + bb), *(const v8h*)(wl + bb + 16),
                                               0, 1, 2, 3, 4, 5, 6, 7, 8, 9, 10, 11, 12, 13, 14, 15);
#pragma unroll
            for (int mi = 0; mi < 4; ++mi) {
                acc[mi][ni] = __builtin_amdgcn_wmma_f32_16x16x32_f16(
                    false, ahi[mi], false, bhi, (short)0, acc[mi][ni], false, false);
                acc[mi][ni] = __builtin_amdgcn_wmma_f32_16x16x32_f16(
                    false, ahi[mi], false, blo, (short)0, acc[mi][ni], false, false);
                acc[mi][ni] = __builtin_amdgcn_wmma_f32_16x16x32_f16(
                    false, alo[mi], false, bhi, (short)0, acc[mi][ni], false, false);
            }
        }
    }
    // C/D layout: VGPR r holds (M = r + hx*8, N = l15)
#pragma unroll
    for (int mi = 0; mi < 4; ++mi)
#pragma unroll
        for (int ni = 0; ni < 4; ++ni) {
            float* zp = z + (size_t)(m0 + mi * 16 + hx * 8) * H_DIM + (n0 + ni * 16 + l15);
#pragma unroll
            for (int r = 0; r < 8; ++r) zp[(size_t)r * H_DIM] = acc[mi][ni][r];
        }
}

// ---------------------------------------------------------------------------
// Top-k radix select, one workgroup (256 threads) per row.  4 rounds of
// 8-bit MSB radix over order keys with an LDS histogram, then a final pass
// that zeroes non-selected entries in place and emits the 50 (idx,val) pairs.
// The 64 KB row stays L2/L0-resident across the 5 passes.
// ---------------------------------------------------------------------------
__global__ __launch_bounds__(256) void sae_topk(
        float* __restrict__ z, int* __restrict__ sel_idx, float* __restrict__ sel_val) {
    const int row = blockIdx.x;
    float* zr = z + (size_t)row * H_DIM;
    const int t = threadIdx.x;

    __shared__ unsigned hist[256];
    __shared__ unsigned sh_prefix;
    __shared__ int      sh_need;
    __shared__ unsigned sh_gt, sh_eq;

    unsigned prefix = 0, prefmask = 0;
    int need = TOPK;

    for (int r = 0; r < 4; ++r) {
        const int shift = 24 - 8 * r;
        hist[t] = 0;
        __syncthreads();
#pragma unroll 4
        for (int i = 0; i < H_DIM / 256; ++i) {
            unsigned key = ord_key(zr[t + i * 256]);
            if ((key & prefmask) == prefix)
                atomicAdd(&hist[(key >> shift) & 255u], 1u);
        }
        __syncthreads();
        if (t == 0) {
            int cum = 0, b = 255;
            for (; b > 0; --b) {
                int c = (int)hist[b];
                if (cum + c >= need) break;
                cum += c;
            }
            sh_prefix = prefix | ((unsigned)b << shift);
            sh_need   = need - cum;
        }
        __syncthreads();
        prefix = sh_prefix;
        need   = sh_need;
        prefmask |= (0xFFu << shift);
        __syncthreads();
    }

    const unsigned T = prefix;          // key of the 50th-largest element
    const int n_gt = TOPK - need;       // count of keys strictly > T
    if (t == 0) { sh_gt = 0; sh_eq = 0; }
    __syncthreads();

#pragma unroll 4
    for (int i = 0; i < H_DIM / 256; ++i) {
        const int col = t + i * 256;
        const float v = zr[col];
        const unsigned key = ord_key(v);
        float outv = 0.0f;
        if (key > T) {
            unsigned s = atomicAdd(&sh_gt, 1u);
            outv = v;
            sel_idx[(size_t)row * TOPK + s] = col;
            sel_val[(size_t)row * TOPK + s] = v;
        } else if (key == T) {
            unsigned s = atomicAdd(&sh_eq, 1u);
            if ((int)s < need) {
                outv = v;
                sel_idx[(size_t)row * TOPK + n_gt + s] = col;
                sel_val[(size_t)row * TOPK + n_gt + s] = v;
            }
        }
        zr[col] = outv;                 // sparse z written in place
    }
}

// ---------------------------------------------------------------------------
// W_dec [1024,16384] -> W_dec_T [16384,1024] so the decoder gather is
// row-contiguous (64 MB, L2-resident on the 192 MB L2).
// ---------------------------------------------------------------------------
__global__ __launch_bounds__(256) void sae_transpose(
        const float* __restrict__ Wd, float* __restrict__ WdT) {
    __shared__ float tile[32][33];
    const int bx = blockIdx.x;          // along H (512 tiles)
    const int by = blockIdx.y;          // along D (32 tiles)
    const int tx = threadIdx.x & 31;
    const int ty = threadIdx.x >> 5;    // 8 row-groups
#pragma unroll
    for (int r = ty; r < 32; r += 8)
        tile[r][tx] = Wd[(size_t)(by * 32 + r) * H_DIM + bx * 32 + tx];
    __syncthreads();
#pragma unroll
    for (int r = ty; r < 32; r += 8)
        WdT[(size_t)(bx * 32 + r) * K_DIM + by * 32 + tx] = tile[tx][r];
}

// ---------------------------------------------------------------------------
// Decoder: recon[b,:] = sum_j val_j * W_dec_T[idx_j,:].  One WG per row,
// each thread owns 4 contiguous outputs (float4 loads/stores).
// ---------------------------------------------------------------------------
__global__ __launch_bounds__(256) void sae_decode(
        const float* __restrict__ WdT, const int* __restrict__ sel_idx,
        const float* __restrict__ sel_val, float* __restrict__ recon) {
    const int row = blockIdx.x;
    const int t = threadIdx.x;
    __shared__ int   sidx[TOPK];
    __shared__ float sval[TOPK];
    if (t < TOPK) {
        sidx[t] = sel_idx[(size_t)row * TOPK + t];
        sval[t] = sel_val[(size_t)row * TOPK + t];
    }
    __syncthreads();
    float4 acc = {0.f, 0.f, 0.f, 0.f};
#pragma unroll 5
    for (int j = 0; j < TOPK; ++j) {
        const float4 w = *(const float4*)(WdT + (size_t)sidx[j] * K_DIM + t * 4);
        const float v = sval[j];
        acc.x += v * w.x; acc.y += v * w.y; acc.z += v * w.z; acc.w += v * w.w;
    }
    *(float4*)(recon + (size_t)row * K_DIM + t * 4) = acc;
}

// ---------------------------------------------------------------------------
extern "C" void kernel_launch(void* const* d_in, const int* in_sizes, int n_in,
                              void* d_out, int out_size, void* d_ws, size_t ws_size,
                              hipStream_t stream) {
    const float* x    = (const float*)d_in[0];   // [8192,1024]
    const float* Wenc = (const float*)d_in[1];   // [16384,1024]
    const float* Wdec = (const float*)d_in[2];   // [1024,16384]

    float* recon = (float*)d_out;                          // [8192,1024]
    float* z     = (float*)d_out + (size_t)B_ROWS * K_DIM; // [8192,16384]

    // workspace layout (all offsets 32B-aligned)
    char* ws = (char*)d_ws;
    float*    WdT     = (float*)ws;                               // 64 MB
    ws += (size_t)H_DIM * K_DIM * sizeof(float);
    int*      sel_idx = (int*)ws;                                 // 1.6 MB
    ws += (size_t)B_ROWS * TOPK * sizeof(int);
    float*    sel_val = (float*)ws;                               // 1.6 MB
    ws += (size_t)B_ROWS * TOPK * sizeof(float);
    _Float16* x_hi    = (_Float16*)ws;                            // 16 MB
    ws += (size_t)B_ROWS * K_DIM * sizeof(_Float16);
    _Float16* x_lo    = (_Float16*)ws;                            // 16 MB
    ws += (size_t)B_ROWS * K_DIM * sizeof(_Float16);
    _Float16* we_hi   = (_Float16*)ws;                            // 32 MB
    ws += (size_t)H_DIM * K_DIM * sizeof(_Float16);
    _Float16* we_lo   = (_Float16*)ws;                            // 32 MB

    // 1) one-shot f16 hi/lo decomposition of x and W_enc
    const int nx8 = B_ROWS * K_DIM / 8;   // 1,048,576
    const int nw8 = H_DIM * K_DIM / 8;    // 2,097,152
    sae_split_f16<<<nx8 / 256, 256, 0, stream>>>(x, x_hi, x_lo, nx8);
    sae_split_f16<<<nw8 / 256, 256, 0, stream>>>(Wenc, we_hi, we_lo, nw8);

    // 2) transpose W_dec for contiguous decoder gathers
    sae_transpose<<<dim3(H_DIM / 32, K_DIM / 32), 256, 0, stream>>>(Wdec, WdT);

    // 3) encoder GEMM: 128 x 256 wave-tiles of 64x64, 8 waves per block
    sae_enc_gemm<<<(B_ROWS / 64) * (H_DIM / 64) / 8, 256, 0, stream>>>(
        x_hi, x_lo, we_hi, we_lo, z);

    // 4) per-row top-50 radix select; zeroes z in place, emits (idx,val) pairs
    sae_topk<<<B_ROWS, 256, 0, stream>>>(z, sel_idx, sel_val);

    // 5) sparse decode
    sae_decode<<<B_ROWS, 256, 0, stream>>>(WdT, sel_idx, sel_val, recon);
}
